// DeformConv2d_75754633167014
// MI455X (gfx1250) — compile-verified, matrix operand order checked by reading, not run on hardware
//
#include <hip/hip_runtime.h>

#define B_ 8
#define C_ 64
#define O_ 64
#define H_ 128
#define W_ 128
#define HW_ (H_*W_)
#define KK_ 9
#define KTOT (C_*KK_)    // 576 (logical K)
#define STR  648         // padded LDS row stride (elems): 1296 B = 324 dwords, 324%64=4
#define TILE_M 32
#define THREADS 256

#define SW_PAD_BYTES  (O_*STR*2)      // 82944
#define WOM_PAD_BYTES (48*STR*2)      // 62208

typedef __attribute__((ext_vector_type(16))) __bf16 v16bf;
typedef __attribute__((ext_vector_type(8)))  __bf16 v8bf;
typedef __attribute__((ext_vector_type(8)))  float  v8f;
typedef unsigned int u32x4 __attribute__((ext_vector_type(4)));
typedef int          i32x4 __attribute__((ext_vector_type(4)));
typedef int          i32x8 __attribute__((ext_vector_type(8)));

#if defined(__has_include)
#  if __has_include(<hip/amd_detail/amd_gfx1250_TDM.h>)
#    define TDM_6ARG 1
#  endif
#endif

__device__ __forceinline__ unsigned short f2bf(float f) {
  unsigned int u = __float_as_uint(f);
  u += 0x7FFFu + ((u >> 16) & 1u);     // round-to-nearest-even
  return (unsigned short)(u >> 16);
}

// 1-D TDM copy: global (bf16, pre-packed) -> LDS. nBytes % 8 == 0, nBytes/8 < 65536.
__device__ __forceinline__ void tdm_load_1d(unsigned ldsOff, const void* gsrc,
                                            unsigned nBytes) {
  const unsigned n = nBytes >> 3;                       // 8-byte elements
  const unsigned long long ga = (unsigned long long)(uintptr_t)gsrc;
  u32x4 g0;
  g0[0] = 1u;                                           // count=1, user mode
  g0[1] = ldsOff;                                       // lds_addr
  g0[2] = (unsigned)ga;                                 // global_addr[31:0]
  g0[3] = (unsigned)((ga >> 32) & 0x01FFFFFFull)        // global_addr[56:32]
        | (2u << 30);                                   // type = 2 ("image")
  i32x8 g1;
  g1[0] = (int)(3u << 16);                              // data_size=3 (8B), wg_mask=0
  g1[1] = (int)((n & 0xFFFFu) << 16);                   // tensor_dim0[15:0]
  g1[2] = (int)((n >> 16) & 0xFFFFu);                   // tensor_dim0[31:16]
  g1[3] = (int)((n & 0xFFFFu) << 16);                   // tile_dim0 = n
  g1[4] = 0;                                            // tile_dim1=0 (1-D), tile_dim2=0
  g1[5] = (int)n;                                       // tensor_dim0_stride[31:0]
  g1[6] = 0;
  g1[7] = 0;
  i32x4 gz4 = {0, 0, 0, 0};
#if defined(TDM_6ARG)
  i32x8 gz8 = {0, 0, 0, 0, 0, 0, 0, 0};
  __builtin_amdgcn_tensor_load_to_lds(g0, g1, gz4, gz4, gz8, 0);
#else
  __builtin_amdgcn_tensor_load_to_lds(g0, g1, gz4, gz4, 0);
#endif
}

struct Rec { int cy0, cy1, cx0, cx1; float w00, w01, w10, w11; };

// One 16x16 tile: A-rows/B-columns live in padded-stride LDS rows; 18 bf16 WMMAs.
__device__ __forceinline__ v8f gemm576(const unsigned short* __restrict__ aRow,
                                       const unsigned short* __restrict__ bRow,
                                       int lhalf) {
  v8f acc = {};
  #pragma unroll
  for (int kb = 0; kb < KTOT/32; ++kb) {
    const int k0 = kb*32 + lhalf*8;
    v8bf alo = *reinterpret_cast<const v8bf*>(aRow + k0);
    v8bf ahi = *reinterpret_cast<const v8bf*>(aRow + k0 + 16);
    v8bf blo = *reinterpret_cast<const v8bf*>(bRow + k0);
    v8bf bhi = *reinterpret_cast<const v8bf*>(bRow + k0 + 16);
    v16bf af, bf;
    #pragma unroll
    for (int i = 0; i < 8; ++i) {
      af[i] = alo[i]; af[i+8] = ahi[i];
      bf[i] = blo[i]; bf[i+8] = bhi[i];
    }
    acc = __builtin_amdgcn_wmma_f32_16x16x32_bf16(
        false, af, false, bf, (short)0, acc, false, false);
  }
  return acc;
}

// ---- Prep: convert conv weights to bf16, padded-stride WMMA layouts, once ----
__global__ __launch_bounds__(THREADS)
void prep_weights(const float* __restrict__ w_off, const float* __restrict__ w_mod,
                  const float* __restrict__ w_reg,
                  unsigned short* __restrict__ ws_sW,
                  unsigned short* __restrict__ ws_wom) {
  const int i = blockIdx.x*THREADS + threadIdx.x;
  if (i < O_*KTOT) {                       // sW[o][k*64+c] = w_reg[o][c][k]
    const int o = i / KTOT, rem = i - o*KTOT;
    const int k = rem / C_,  c  = rem - k*C_;
    ws_sW[o*STR + k*C_ + c] = f2bf(w_reg[(o*C_ + c)*KK_ + k]);
  } else if (i < O_*KTOT + 48*KTOT) {      // wom rows: 0..17 w_off | pad | 32..40 w_mod | pad
    const int i2 = i - O_*KTOT;
    const int j = i2 / KTOT, rem = i2 - j*KTOT;
    const int k = rem / C_,  c  = rem - k*C_;
    float v = 0.0f;
    if (j < 18)                 v = w_off[(j*C_ + c)*KK_ + k];
    else if (j >= 32 && j < 41) v = w_mod[((j-32)*C_ + c)*KK_ + k];
    ws_wom[j*STR + k*C_ + c] = f2bf(v);
  }
}

__global__ __launch_bounds__(THREADS, 1)
void deform_conv_fused(const float* __restrict__ x, const float* __restrict__ z,
                       const float* __restrict__ b_off, const float* __restrict__ b_mod,
                       const unsigned short* __restrict__ ws_sW,
                       const unsigned short* __restrict__ ws_wom,
                       float* __restrict__ out) {
  // Overlaid LDS arena (rows padded to STR elems; stride%256B==16 -> conflict-free b128):
  //  stage 1: zIm[32xSTR] | xIm[32xSTR] | wom[48xSTR]   (145152 B)
  //  stage 2: sW[64xSTR]  | sA[32xSTR]                  (124416 B)
  __shared__ __align__(16) unsigned char smemRaw[145152];
  unsigned short* zIm = (unsigned short*)smemRaw;              // 41472 B
  unsigned short* xIm = (unsigned short*)(smemRaw + 41472);    // 41472 B
  unsigned short* wom = (unsigned short*)(smemRaw + 82944);    // 62208 B
  unsigned short* sW  = (unsigned short*)smemRaw;              // 82944 B (stage 2)
  unsigned short* sA  = (unsigned short*)(smemRaw + 82944);    // 41472 B (stage 2)

  __shared__ float sOff[TILE_M][18];
  __shared__ float sMask[TILE_M][KK_];
  __shared__ Rec   sRec[TILE_M*KK_];

  const int tid   = threadIdx.x;
  const int lane  = tid & 31;
  const int wave  = tid >> 5;
  const int lrow  = lane & 15;
  const int lhalf = lane >> 4;
  const int c     = tid & 63;           // channel owned by this thread
  const int g     = tid >> 6;           // 0..3

  const int p0 = blockIdx.x * TILE_M;   // 32 consecutive pixels: same (b, h)
  const int b  = p0 / HW_;
  const int hw = p0 % HW_;
  const int h  = hw / W_;
  const int w0 = hw % W_;

  // Async-stage offset/mask weights into LDS via the Tensor Data Mover,
  // overlapped with the im2col build below.
  if (wave == 7)
    tdm_load_1d((unsigned)(uintptr_t)wom, ws_wom, WOM_PAD_BYTES);

  // Warm L2/L1 for the main-GEMM weight block consumed after phase 1.
  for (int off = tid*256; off < SW_PAD_BYTES; off += THREADS*256)
    __builtin_prefetch((const char*)ws_sW + off, 0, 1);

  // ---- Stage 1: z/x im2col tiles in bf16 (zero-pad borders) ----
  for (int pk = g; pk < TILE_M*KK_; pk += 4) {
    const int p  = pk / KK_;
    const int k  = pk - p*KK_;
    const int yy = h + (k/3) - 1;
    const int xx = w0 + p + (k%3) - 1;
    unsigned short zv = 0, xv = 0;
    if (yy >= 0 && yy < H_ && xx >= 0 && xx < W_) {
      const size_t idx = ((size_t)b*C_ + c)*HW_ + (size_t)yy*W_ + xx;
      zv = f2bf(z[idx]);
      xv = f2bf(x[idx]);
    }
    zIm[p*STR + k*C_ + c] = zv;
    xIm[p*STR + k*C_ + c] = xv;
  }
  if (wave == 7)
    __builtin_amdgcn_s_wait_tensorcnt(0);
  __syncthreads();

  // ---- Phase 1: offset + mask convs as WMMA GEMMs (waves 0..5) ----
  if (wave < 6) {
    int mt, colBase, isMask;
    const unsigned short *aBase, *bBase;
    if (wave < 4) { mt = wave >> 1; colBase = (wave & 1)*16; isMask = 0;
                    aBase = zIm; bBase = wom + colBase*STR; }
    else          { mt = wave - 4; colBase = 0;              isMask = 1;
                    aBase = xIm; bBase = wom + 32*STR; }
    v8f acc = gemm576(aBase + (mt*16 + lrow)*STR, bBase + lrow*STR, lhalf);
    const int col = colBase + lrow;     // conv output channel
    #pragma unroll
    for (int r = 0; r < 8; ++r) {
      const int p = mt*16 + lhalf*8 + r;
      if (!isMask) {
        if (col < 18) sOff[p][col] = acc[r] + b_off[col];
      } else {
        if (col < 9)  sMask[p][col] = 2.0f / (1.0f + __expf(-(acc[r] + b_mod[col])));
      }
    }
  }
  __syncthreads();   // zIm/xIm/wom dead beyond this point

  // Async-stage main GEMM weights into the (now free) sW region, overlapped
  // with record building + deformed gathers below.
  if (wave == 6)
    tdm_load_1d((unsigned)(uintptr_t)sW, ws_sW, SW_PAD_BYTES);

  // ---- Phase 1b: bilinear corner records per (pixel, tap); mask folded in ----
  for (int t = tid; t < TILE_M*KK_; t += THREADS) {
    const int p = t / KK_;
    const int k = t - p*KK_;
    const float dy = sOff[p][2*k];
    const float dx = sOff[p][2*k+1];
    const float m  = sMask[p][k];
    const float py = dy + (float)(k/3) + (float)(h - 1);
    const float px = dx + (float)(k%3) + (float)(w0 + p - 1);
    const float y0f = floorf(py), x0f = floorf(px);
    const float wy1 = py - y0f, wy0 = 1.0f - wy1;
    const float wx1 = px - x0f, wx0 = 1.0f - wx1;
    const int y0 = (int)y0f, x0i = (int)x0f;
    const int y1 = y0 + 1,   x1i = x0i + 1;
    const float vy0 = (y0  >= 0 && y0  < H_) ? 1.0f : 0.0f;
    const float vy1 = (y1  >= 0 && y1  < H_) ? 1.0f : 0.0f;
    const float vx0 = (x0i >= 0 && x0i < W_) ? 1.0f : 0.0f;
    const float vx1 = (x1i >= 0 && x1i < W_) ? 1.0f : 0.0f;
    Rec r;
    r.cy0 = min(max(y0, 0), H_-1);
    r.cy1 = min(max(y1, 0), H_-1);
    r.cx0 = min(max(x0i, 0), W_-1);
    r.cx1 = min(max(x1i, 0), W_-1);
    r.w00 = m*wy0*wx0*vy0*vx0;
    r.w01 = m*wy0*wx1*vy0*vx1;
    r.w10 = m*wy1*wx0*vy1*vx0;
    r.w11 = m*wy1*wx1*vy1*vx1;
    sRec[t] = r;
  }
  __syncthreads();

  // ---- Stage 2: build modulated im2col A (bf16) while TDM fills sW ----
  {
    const float* xb = x + ((size_t)b*C_ + c)*HW_;
    for (int pk = g; pk < TILE_M*KK_; pk += 4) {
      const Rec r = sRec[pk];
      const int p = pk / KK_;
      const int k = pk - p*KK_;
      const float v = r.w00 * xb[r.cy0*W_ + r.cx0]
                    + r.w01 * xb[r.cy0*W_ + r.cx1]
                    + r.w10 * xb[r.cy1*W_ + r.cx0]
                    + r.w11 * xb[r.cy1*W_ + r.cx1];
      sA[p*STR + k*C_ + c] = f2bf(v);
    }
  }
  if (wave == 6)
    __builtin_amdgcn_s_wait_tensorcnt(0);
  __syncthreads();

  // ---- Phase 3: main WMMA GEMM: out[32 x 64] = A[32x576] * W^T[576x64] ----
  const int mtile = wave >> 2;          // 0..1
  const int ntile = wave & 3;           // 0..3
  v8f acc = gemm576(sA + (mtile*16 + lrow)*STR,
                    sW + (ntile*16 + lrow)*STR, lhalf);

  // D layout: VGPR r, lane<16 -> M=r, lane>=16 -> M=8+r; N = lane%16
  const int o = ntile*16 + lrow;
  float* op = out + (((size_t)b*O_ + o)*H_ + h)*W_ + w0 + mtile*16 + lhalf*8;
  *reinterpret_cast<float4*>(op)     = make_float4(acc[0], acc[1], acc[2], acc[3]);
  *reinterpret_cast<float4*>(op + 4) = make_float4(acc[4], acc[5], acc[6], acc[7]);
}

extern "C" void kernel_launch(void* const* d_in, const int* in_sizes, int n_in,
                              void* d_out, int out_size, void* d_ws, size_t ws_size,
                              hipStream_t stream) {
  (void)in_sizes; (void)n_in; (void)out_size; (void)ws_size;
  const float* x     = (const float*)d_in[0];
  const float* z     = (const float*)d_in[1];
  const float* w_off = (const float*)d_in[2];
  const float* b_off = (const float*)d_in[3];
  const float* w_mod = (const float*)d_in[4];
  const float* b_mod = (const float*)d_in[5];
  const float* w_reg = (const float*)d_in[6];
  float* out = (float*)d_out;

  unsigned short* ws_sW  = (unsigned short*)d_ws;                          // 82944 B
  unsigned short* ws_wom = (unsigned short*)((char*)d_ws + SW_PAD_BYTES);  // 62208 B

  const int prepElems  = O_*KTOT + 48*KTOT;                  // 64512
  const int prepBlocks = (prepElems + THREADS - 1) / THREADS;
  prep_weights<<<dim3(prepBlocks), dim3(THREADS), 0, stream>>>(
      w_off, w_mod, w_reg, ws_sW, ws_wom);

  const int nblocks = (B_*H_*W_) / TILE_M;   // 4096
  deform_conv_fused<<<dim3(nblocks), dim3(THREADS), 0, stream>>>(
      x, z, b_off, b_mod, ws_sW, ws_wom, out);
}
